// NeuralCoilLayer_51805895524365
// MI455X (gfx1250) — matrix operand order repaired and checked by reading, not compiled, not run on hardware
//
#include <hip/hip_runtime.h>
#include <hip/hip_bf16.h>
#include <math.h>

typedef __bf16 bf16_t;
typedef __bf16 v16bf __attribute__((ext_vector_type(16)));
typedef __bf16 bf2   __attribute__((ext_vector_type(2)));
typedef float  v8f   __attribute__((ext_vector_type(8)));

#define BATCH 32
#define SEQL  128
#define FD    64
#define KD    65            // F+1
#define IJ    4096          // F*F (flattened i*64+j)
#define NTILE 256           // IJ/16
#define FRAG_PER_K 131072   // NTILE*2(bt)*32(lane)*8(r) elements per k slice

// Fragment-layout index helper (documenting the mapping):
//   element(k, it, bt, lane, r) = ((((k*256+it)*2+bt)*32 + lane)*8 + r)
//   -> ij = it*16 + (lane<16 ? 0 : 8) + r ;  b = bt*16 + (lane&15)

// ---------------------------------------------------------------------------
// K0: one-time pack of interaction[F,F,F,F+1] (fp32) into bf16 WMMA A-fragment
// layout. Fragment f = ((k*256 + it)*2 + ks); within fragment, lane loads 16
// contiguous bf16 (32B). A-matrix 16x32 bf16 layout: lane = M%16 group,
// lanes 0-15 hold K={0..7,16..23}, lanes 16-31 hold K={8..15,24..31}.
// ---------------------------------------------------------------------------
__global__ void pack_interaction(const float* __restrict__ inter,
                                 bf16_t* __restrict__ apack) {
    long d = (long)blockIdx.x * blockDim.x + threadIdx.x;   // dword index
    if (d >= 8519680L) return;                              // 65*256*2*32*8
    int v    = (int)(d & 7);
    int lane = (int)((d >> 3) & 31);
    long f   = d >> 8;                                      // fragment id
    int ks   = (int)(f & 1);
    int it   = (int)((f >> 1) & 255);
    int k    = (int)(f >> 9);

    int M  = lane & 15;
    int ij = it * 16 + M;
    int base8 = (lane < 16) ? 0 : 8;
    int K0 = (v < 4) ? (2 * v + base8) : (16 + 2 * (v - 4) + base8);
    int m0 = ks * 32 + K0;

    const float* src = inter + ((long)ij * FD + m0) * KD + k;
    bf2 p;
    p.x = (bf16_t)src[0];
    p.y = (bf16_t)src[KD];      // m0+1 -> stride KD
    ((bf2*)apack)[d] = p;
}

// ---------------------------------------------------------------------------
// Init T0 = 1/F (softmax of zeros), written into tail of d_out.
// ---------------------------------------------------------------------------
__global__ void init_T(float* __restrict__ T) {
    int i = blockIdx.x * 256 + threadIdx.x;
    if (i < BATCH * FD * FD) T[i] = 1.0f / FD;
}

// ---------------------------------------------------------------------------
// K1: per-step softmaxes hi = softmax(s*1000), lo = softmax(1 - s*1000) and
// weight table wT_frag (fragment layout, no k dim) = lo[b][i]*hi[b][j].
// One block per batch b, 64 threads (thread = column j).
// ---------------------------------------------------------------------------
__global__ void prep_softmax(const float* __restrict__ x, int t,
                             float* __restrict__ wTf) {
    __shared__ float hiS[FD], loS[FD], red[2];
    int b = blockIdx.x, tid = threadIdx.x;
    float s = x[((long)b * SEQL + t) * FD + tid];
    hiS[tid] = s * 1000.0f;
    loS[tid] = 1.0f - s * 1000.0f;
    __syncthreads();
    if (tid == 0) {
        float m1 = -1e30f, m2 = -1e30f;
        for (int i = 0; i < FD; i++) {
            if (hiS[i] > m1) m1 = hiS[i];
            if (loS[i] > m2) m2 = loS[i];
        }
        red[0] = m1; red[1] = m2;
    }
    __syncthreads();
    float eh = __expf(s * 1000.0f - red[0]);
    float el = __expf((1.0f - s * 1000.0f) - red[1]);
    __syncthreads();
    hiS[tid] = eh; loS[tid] = el;
    __syncthreads();
    if (tid == 0) {
        float s1 = 0.f, s2 = 0.f;
        for (int i = 0; i < FD; i++) { s1 += hiS[i]; s2 += loS[i]; }
        red[0] = 1.0f / s1; red[1] = 1.0f / s2;
    }
    __syncthreads();
    hiS[tid] = eh * red[0];
    loS[tid] = el * red[1];
    __syncthreads();
    float hj = hiS[tid];
    int bt = b >> 4, lnLow = b & 15;
    for (int i = 0; i < FD; i++) {
        int ij = i * FD + tid;
        int it = ij >> 4, M = ij & 15;
        int lh = M >> 3, r = M & 7;
        wTf[(((long)(it * 2 + bt) * 32) + lh * 16 + lnLow) * 8 + r] = loS[i] * hj;
    }
}

// ---------------------------------------------------------------------------
// K2: pack ns[b,m,k] = (k==0 ? s[b,m] : T[b,m,k-1]) into bf16 WMMA B-fragment
// layout. B-matrix 32x16: lane = N%16, lanes 0-15 hold K={0..7,16..23}, lanes 16-31
// hold K={8..15,24..31}, 2 values per dword. Fragment nb = ((k*2+bt)*2+ks).
// ---------------------------------------------------------------------------
__global__ void pack_ns(const float* __restrict__ x, int t,
                        const float* __restrict__ T,
                        bf16_t* __restrict__ nspack) {
    int d = blockIdx.x * 256 + threadIdx.x;     // < 66560 dwords
    int v    = d & 7;
    int lane = (d >> 3) & 31;
    int ks   = (d >> 8) & 1;
    int bt   = (d >> 9) & 1;
    int k    = d >> 10;
    if (k >= KD) return;
    int b = bt * 16 + (lane & 15);
    int m0 = ks * 32 + ((lane < 16) ? 0 : 8) + ((v < 4) ? 2 * v : 16 + 2 * (v - 4));
    float f0, f1;
    if (k == 0) {
        const float* sp = x + ((long)b * SEQL + t) * FD;
        f0 = sp[m0]; f1 = sp[m0 + 1];
    } else {
        f0 = T[((long)b * FD + m0) * FD + (k - 1)];
        f1 = T[((long)b * FD + m0 + 1) * FD + (k - 1)];
    }
    bf2 p; p.x = (bf16_t)f0; p.y = (bf16_t)f1;
    ((bf2*)nspack)[d] = p;
}

// ---------------------------------------------------------------------------
// K3: the big contraction. Per wave: one k slice, 4 ij-tiles of 16 rows.
// D[16 ij x 32 b] = A[16 x 64] * B[64 x 32], K split into 2 WMMA steps,
// N split into 2 tiles -> 4 WMMAs per tile, 16 per wave. Accumulators are
// stored directly in WMMA fragment layout -> 32B vector stores per acc.
// ---------------------------------------------------------------------------
__global__ void cand_gemm(const bf16_t* __restrict__ apack,
                          const bf16_t* __restrict__ nspack,
                          float* __restrict__ candf) {
    int lane = threadIdx.x & 31;
    int wave = threadIdx.x >> 5;
    int gw   = blockIdx.x * 8 + wave;            // 0..4159
    int k    = gw >> 6;
    int itg  = gw & 63;
    if (k >= KD) return;

    const v16bf* nsv = (const v16bf*)nspack;
    int kb = k * 4;
    v16bf b00 = nsv[(kb + 0) * 32 + lane];       // bt0, ks0
    v16bf b01 = nsv[(kb + 1) * 32 + lane];       // bt0, ks1
    v16bf b10 = nsv[(kb + 2) * 32 + lane];       // bt1, ks0
    v16bf b11 = nsv[(kb + 3) * 32 + lane];       // bt1, ks1

    const v16bf* av = (const v16bf*)apack;

    for (int tt = 0; tt < 4; ++tt) {
        int it = itg * 4 + tt;
        long f = ((long)k * NTILE + it) * 2;
        v16bf a0 = av[(f + 0) * 32 + lane];
        v16bf a1 = av[(f + 1) * 32 + lane];
        v8f acc0 = {}; v8f acc1 = {};
        acc0 = __builtin_amdgcn_wmma_f32_16x16x32_bf16(false, a0, false, b00,
                                                       (short)0, acc0, false, false);
        acc0 = __builtin_amdgcn_wmma_f32_16x16x32_bf16(false, a1, false, b01,
                                                       (short)0, acc0, false, false);
        acc1 = __builtin_amdgcn_wmma_f32_16x16x32_bf16(false, a0, false, b10,
                                                       (short)0, acc1, false, false);
        acc1 = __builtin_amdgcn_wmma_f32_16x16x32_bf16(false, a1, false, b11,
                                                       (short)0, acc1, false, false);
        // fragment-layout stores: 32B vector per accumulator
        long base = (((long)k * NTILE + it) * 2);
        *(v8f*)(candf + ((base + 0) * 32 + lane) * 8) = acc0;
        *(v8f*)(candf + ((base + 1) * 32 + lane) * 8) = acc1;
    }
}

// ---------------------------------------------------------------------------
// K4: scores[b][k] = sum_ij cand[k][ij][b] * w[b][ij], both in fragment
// layout -> fully contiguous streaming loads. One block per k.
// Within a k slice, element i: bt=(i>>8)&1 alternates with chunk index.
// ---------------------------------------------------------------------------
__global__ void scores_kernel(const float* __restrict__ candf,
                              const float* __restrict__ wTf,
                              float* __restrict__ scores) {
    __shared__ float sm0[256], sm1[256];
    int k = blockIdx.x, tid = threadIdx.x;
    const float* ck = candf + (long)k * FRAG_PER_K;
    float a0 = 0.f, a1 = 0.f;
    for (int n = 0; n < FRAG_PER_K / 256; n += 2) {
        a0 += ck[(long)n * 256 + tid]       * wTf[(long)n * 256 + tid];
        a1 += ck[(long)(n + 1) * 256 + tid] * wTf[(long)(n + 1) * 256 + tid];
    }
    sm0[tid] = a0;
    sm1[tid] = a1;
    __syncthreads();
    if (tid < 32) {
        int bt = tid >> 4, lnLow = tid & 15;
        const float* sm = bt ? sm1 : sm0;
        float s = 0.f;
        for (int lh = 0; lh < 2; ++lh)
            for (int r = 0; r < 8; ++r)
                s += sm[(lh * 16 + lnLow) * 8 + r];
        scores[(bt * 16 + lnLow) * KD + k] = s;   // [b][k]
    }
}

// ---------------------------------------------------------------------------
// K5: sw[b][k] = softmax_k(scores[b][k]*1000). 32 threads, one per batch.
// ---------------------------------------------------------------------------
__global__ void softmax_k(const float* __restrict__ scores,
                          float* __restrict__ sw) {
    int b = threadIdx.x;
    float m = -1e30f;
    for (int k = 0; k < KD; k++) {
        float v = scores[b * KD + k] * 1000.0f;
        if (v > m) m = v;
    }
    float s = 0.f;
    for (int k = 0; k < KD; k++) {
        float e = __expf(scores[b * KD + k] * 1000.0f - m);
        s += e;
        sw[b * KD + k] = e;
    }
    float inv = 1.0f / s;
    for (int k = 0; k < KD; k++) sw[b * KD + k] *= inv;
}

// ---------------------------------------------------------------------------
// K6: selpre_frag = sum_k cand_frag * sw[b][k]. One block per (it,bt)
// fragment column; each per-k load is 1KB contiguous. Output stays in
// fragment layout.
// ---------------------------------------------------------------------------
__global__ void selpre_kernel(const float* __restrict__ candf,
                              const float* __restrict__ sw,
                              float* __restrict__ selpref) {
    __shared__ float swL[BATCH * KD];
    int tid = threadIdx.x;
    for (int i = tid; i < BATCH * KD; i += 256) swL[i] = sw[i];
    __syncthreads();
    long base = (long)blockIdx.x * 256;           // (it*2+bt)*256
    int lane = tid >> 3;
    int bt = blockIdx.x & 1;
    int b  = bt * 16 + (lane & 15);
    float acc = 0.f;
    for (int k = 0; k < KD; k++)
        acc += candf[(long)k * FRAG_PER_K + base + tid] * swL[b * KD + k];
    selpref[base + tid] = acc;
}

// ---------------------------------------------------------------------------
// K7: per batch b: sel = softmax_i(selpre*7), T <- sel (carry),
// out[b][t][i] = sum_j sel[i][j]*s[b][j]. One block per b, 256 threads.
// ---------------------------------------------------------------------------
__global__ void rownorm_kernel(const float* __restrict__ selpref,
                               const float* __restrict__ x, int t,
                               float* __restrict__ T,
                               float* __restrict__ out) {
    __shared__ float M[FD * (FD + 1)];    // [i][j], stride 65 to dodge banks
    __shared__ float sL[FD];
    __shared__ float red[256];
    int b = blockIdx.x, tid = threadIdx.x;
    int bt = b >> 4, lnLow = b & 15;

    for (int idx = tid; idx < IJ; idx += 256) {
        int i = idx >> 6, j = idx & 63;
        int it = idx >> 4, Mm = idx & 15;
        int lh = Mm >> 3, r = Mm & 7;
        float v = selpref[(((long)(it * 2 + bt) * 32) + lh * 16 + lnLow) * 8 + r];
        M[i * (FD + 1) + j] = v * 7.0f;
    }
    if (tid < FD) sL[tid] = x[((long)b * SEQL + t) * FD + tid];
    __syncthreads();

    int j = tid & 63, ig = tid >> 6;      // 4 groups of 16 rows per column
    float mx = -1e30f;
    for (int r = 0; r < 16; r++) {
        float v = M[(ig * 16 + r) * (FD + 1) + j];
        if (v > mx) mx = v;
    }
    red[tid] = mx;
    __syncthreads();
    if (ig == 0) {
        float a = red[j];
        for (int g = 1; g < 4; g++) { float v = red[g * 64 + j]; if (v > a) a = v; }
        red[j] = a;
    }
    __syncthreads();
    float cmax = red[j];
    __syncthreads();                      // done reading red before reuse

    float ss = 0.f;
    for (int r = 0; r < 16; r++) {
        int i = ig * 16 + r;
        float e = __expf(M[i * (FD + 1) + j] - cmax);
        M[i * (FD + 1) + j] = e;
        ss += e;
    }
    red[tid] = ss;
    __syncthreads();
    if (ig == 0) {
        float a = 0.f;
        for (int g = 0; g < 4; g++) a += red[g * 64 + j];
        red[j] = 1.0f / a;
    }
    __syncthreads();
    float inv = red[j];
    for (int r = 0; r < 16; r++) {
        int i = ig * 16 + r;
        M[i * (FD + 1) + j] *= inv;
    }
    __syncthreads();

    // T carry (coalesced)
    for (int idx = tid; idx < IJ; idx += 256) {
        int i = idx >> 6, jj = idx & 63;
        T[((long)b * FD + i) * FD + jj] = M[i * (FD + 1) + jj];
    }

    // new_s[b][i] = sum_j sel[i][j] * s[b][j]
    int i2 = tid & 63, jg = tid >> 6;
    float acc = 0.f;
    for (int r = 0; r < 16; r++) {
        int jj = jg * 16 + r;
        acc += M[i2 * (FD + 1) + jj] * sL[jj];
    }
    red[tid] = acc;
    __syncthreads();
    if (jg == 0) {
        float a = 0.f;
        for (int g = 0; g < 4; g++) a += red[g * 64 + i2];
        out[((long)b * SEQL + t) * FD + i2] = a;
    }
}

// ---------------------------------------------------------------------------
extern "C" void kernel_launch(void* const* d_in, const int* in_sizes, int n_in,
                              void* d_out, int out_size, void* d_ws, size_t ws_size,
                              hipStream_t stream) {
    (void)in_sizes; (void)n_in; (void)out_size; (void)ws_size;
    const float* x     = (const float*)d_in[0];   // [32,128,64]
    const float* inter = (const float*)d_in[1];   // [64,64,64,65]
    float* out = (float*)d_out;                   // [32,128,64] ++ [32,64,64]
    float* T   = out + (long)BATCH * SEQL * FD;   // carry lives in d_out tail

    char* ws = (char*)d_ws;
    size_t off = 0;
    auto wsalloc = [&](size_t bytes) -> void* {
        void* p = ws + off;
        off += (bytes + 255) & ~(size_t)255;
        return p;
    };
    bf16_t* apack  = (bf16_t*)wsalloc((size_t)KD * IJ * FD * 2);        // 34 MB
    float*  candf  = (float*)wsalloc((size_t)KD * FRAG_PER_K * 4);      // 34 MB
    bf16_t* nspack = (bf16_t*)wsalloc((size_t)KD * 4 * 512 * 2);        // 266 KB
    float*  wTf    = (float*)wsalloc((size_t)FRAG_PER_K * 4);           // 524 KB
    float*  scores = (float*)wsalloc((size_t)BATCH * KD * 4);
    float*  sw     = (float*)wsalloc((size_t)BATCH * KD * 4);
    float*  selpref= (float*)wsalloc((size_t)FRAG_PER_K * 4);           // 524 KB

    pack_interaction<<<33280, 256, 0, stream>>>(inter, apack);
    init_T<<<(BATCH * FD * FD + 255) / 256, 256, 0, stream>>>(T);

    for (int t = 0; t < SEQL; ++t) {
        prep_softmax  <<<32,  64, 0, stream>>>(x, t, wTf);
        pack_ns       <<<260, 256, 0, stream>>>(x, t, T, nspack);
        cand_gemm     <<<520, 256, 0, stream>>>(apack, nspack, candf);
        scores_kernel <<<65,  256, 0, stream>>>(candf, wTf, scores);
        softmax_k     <<<1,    32, 0, stream>>>(scores, sw);
        selpre_kernel <<<512, 256, 0, stream>>>(candf, sw, selpref);
        rownorm_kernel<<<32,  256, 0, stream>>>(selpref, x, t, T, out);
    }
}
